// PMLP_APPNP_79353815761145
// MI455X (gfx1250) — compile-verified, matrix operand order checked by reading, not run on hardware
//
#include <hip/hip_runtime.h>
#include <hip/hip_bf16.h>
#include <math.h>

// ---------------------------------------------------------------------------
// PMLP + APPNP for MI455X (gfx1250, wave32, WMMA)
//   h = relu(BN(x@W0+b0)) @ W1 + b1 ;  5x: h = scatter(w*h[src] -> dst) + dinv^2*h
// ---------------------------------------------------------------------------

typedef __bf16 bf16;
typedef bf16  v16bf __attribute__((ext_vector_type(16)));
typedef bf16  v8bf  __attribute__((ext_vector_type(8)));
typedef float v8f   __attribute__((ext_vector_type(8)));

#define IN_C  128
#define HID   64
#define OUT_C 64

// -------------------------- utility kernels --------------------------------

__global__ __launch_bounds__(256) void zero_f32(float* __restrict__ p, int n) {
  int i = blockIdx.x * 256 + threadIdx.x;
  if (i < n) p[i] = 0.0f;
}

// degree of dst over non-self edges (counts accumulated as float)
__global__ __launch_bounds__(256) void deg_kernel(const int* __restrict__ src,
                                                  const int* __restrict__ dst,
                                                  float* __restrict__ deg, int E) {
  int e = blockIdx.x * 256 + threadIdx.x;
  if (e < E) {
    int s = src[e], d = dst[e];
    if (s != d) atomicAdd(&deg[d], 1.0f);
  }
}

// dinv = rsqrt(deg + 1)   (in place over the deg buffer)
__global__ __launch_bounds__(256) void dinv_kernel(float* __restrict__ deg, int n) {
  int i = blockIdx.x * 256 + threadIdx.x;
  if (i < n) deg[i] = rsqrtf(deg[i] + 1.0f);
}

// per-edge normalized weight, 0 for self loops
__global__ __launch_bounds__(256) void edgew_kernel(const int* __restrict__ src,
                                                    const int* __restrict__ dst,
                                                    const float* __restrict__ dinv,
                                                    float* __restrict__ w, int E) {
  int e = blockIdx.x * 256 + threadIdx.x;
  if (e < E) {
    int s = src[e], d = dst[e];
    w[e] = (s != d) ? dinv[s] * dinv[d] : 0.0f;
  }
}

// mean / rstd from accumulated sum & sumsq (biased variance, eps = 1e-10)
__global__ __launch_bounds__(64) void bn_finalize(const float* __restrict__ gsum,
                                                  const float* __restrict__ gsq,
                                                  float* __restrict__ mean,
                                                  float* __restrict__ rstd, int N) {
  int c = threadIdx.x;
  float invN = 1.0f / (float)N;
  float m = gsum[c] * invN;
  float v = gsq[c] * invN - m * m;
  mean[c] = m;
  rstd[c] = rsqrtf(v + 1e-10f);
}

// out[i,:] = dinv[i]^2 * in[i,:]   (float4 vectorized; 16 threads / node)
__global__ __launch_bounds__(256) void prop_init(const float* __restrict__ hin,
                                                 const float* __restrict__ dinv,
                                                 float* __restrict__ hout, int N) {
  int idx = blockIdx.x * 256 + threadIdx.x;
  if (idx < N * 16) {
    int i = idx >> 4, cg = (idx & 15) << 2;
    float sw = dinv[i] * dinv[i];
    const float4 v = *(const float4*)(hin + (size_t)i * 64 + cg);
    float4 o;
    o.x = sw * v.x; o.y = sw * v.y; o.z = sw * v.z; o.w = sw * v.w;
    *(float4*)(hout + (size_t)i * 64 + cg) = o;
  }
}

// out[dst,:] += w[e] * in[src,:]   (16 threads / edge, 4 channels each)
__global__ __launch_bounds__(256) void prop_scatter(const float* __restrict__ hin,
                                                    float* __restrict__ hout,
                                                    const int* __restrict__ src,
                                                    const int* __restrict__ dst,
                                                    const float* __restrict__ w, int E) {
  int idx = blockIdx.x * 256 + threadIdx.x;
  int e = idx >> 4;
  if (e >= E) return;
  float we = w[e];
  if (we == 0.0f) return;
  int cg = (idx & 15) << 2;
  int s = src[e], d = dst[e];
  const float4 v = *(const float4*)(hin + (size_t)s * 64 + cg);
  float* o = hout + (size_t)d * 64 + cg;
  atomicAdd(o + 0, we * v.x);
  atomicAdd(o + 1, we * v.y);
  atomicAdd(o + 2, we * v.z);
  atomicAdd(o + 3, we * v.w);
}

// -------------------------- WMMA GEMM kernels ------------------------------
// Block = 256 threads = 8 waves; block covers 32 rows x 64 cols of output:
// wave (r,c) in 2x4 grid computes one 16x16 tile via v_wmma_f32_16x16x32_bf16.

// h1 = x @ W0 + b0, plus per-column sum / sumsq accumulation for BatchNorm.
__global__ __launch_bounds__(256)
void gemm1_wmma(const float* __restrict__ x, const float* __restrict__ W0,
                const float* __restrict__ b0, float* __restrict__ h1,
                float* __restrict__ gsum, float* __restrict__ gsq, int N) {
  __shared__ __align__(16) bf16 sW[HID * IN_C];  // transposed: sW[n*128 + k]
  __shared__ float sSum[HID], sSq[HID];
  int t = threadIdx.x;
  for (int idx = t; idx < IN_C * HID; idx += 256) {
    int k = idx >> 6, n = idx & 63;
    sW[n * IN_C + k] = (bf16)W0[idx];
  }
  if (t < HID) { sSum[t] = 0.0f; sSq[t] = 0.0f; }
  __syncthreads();

  int wave = t >> 5, lane = t & 31;
  int tile_m = blockIdx.x * 32 + (wave >> 2) * 16;
  int tile_n = (wave & 3) * 16;
  int hi = lane >> 4;       // half-wave selector
  int lr = lane & 15;
  int row = tile_m + lr;    // A-fragment row (same for both halves)
  int col = tile_n + lr;    // B column / C column
  int kb  = hi * 8;

  v8f c = {0.f, 0.f, 0.f, 0.f, 0.f, 0.f, 0.f, 0.f};
  #pragma unroll
  for (int kk = 0; kk < 4; ++kk) {           // K = 128 in 4 steps of 32
    int kbase = kk * 32 + kb;
    const float* xr = x + (size_t)row * IN_C + kbase;
    float4 f0 = *(const float4*)(xr);
    float4 f1 = *(const float4*)(xr + 4);
    float4 f2 = *(const float4*)(xr + 16);
    float4 f3 = *(const float4*)(xr + 20);
    v16bf a;
    a[0]=(bf16)f0.x;  a[1]=(bf16)f0.y;  a[2]=(bf16)f0.z;  a[3]=(bf16)f0.w;
    a[4]=(bf16)f1.x;  a[5]=(bf16)f1.y;  a[6]=(bf16)f1.z;  a[7]=(bf16)f1.w;
    a[8]=(bf16)f2.x;  a[9]=(bf16)f2.y;  a[10]=(bf16)f2.z; a[11]=(bf16)f2.w;
    a[12]=(bf16)f3.x; a[13]=(bf16)f3.y; a[14]=(bf16)f3.z; a[15]=(bf16)f3.w;
    const bf16* wc = &sW[col * IN_C + kbase];
    v8bf blo = *(const v8bf*)(wc);
    v8bf bhi = *(const v8bf*)(wc + 16);
    v16bf b = __builtin_shufflevector(blo, bhi, 0,1,2,3,4,5,6,7,8,9,10,11,12,13,14,15);
    c = __builtin_amdgcn_wmma_f32_16x16x32_bf16(false, a, false, b, (short)0, c,
                                                false, false);
  }

  float bias = b0[col];
  float lsum = 0.0f, lsq = 0.0f;
  #pragma unroll
  for (int v = 0; v < 8; ++v) {
    float hval = c[v] + bias;
    h1[(size_t)(tile_m + hi * 8 + v) * HID + col] = hval;
    lsum += hval; lsq += hval * hval;
  }
  atomicAdd(&sSum[col], lsum);   // ds_add_f32 block reduction
  atomicAdd(&sSq[col], lsq);
  __syncthreads();
  if (t < HID)            atomicAdd(&gsum[t], sSum[t]);
  else if (t < 2 * HID)   atomicAdd(&gsq[t - HID], sSq[t - HID]);
}

// hout = relu((h1 - mean) * rstd) @ W1 + b1   (BN+ReLU fused on A-frag load)
__global__ __launch_bounds__(256)
void gemm2_wmma(const float* __restrict__ h1, const float* __restrict__ W1,
                const float* __restrict__ b1, const float* __restrict__ mean,
                const float* __restrict__ rstd, float* __restrict__ hout, int N) {
  __shared__ __align__(16) bf16 sW[OUT_C * HID];  // transposed: sW[n*64 + k]
  __shared__ float sMean[HID], sRstd[HID];
  int t = threadIdx.x;
  for (int idx = t; idx < HID * OUT_C; idx += 256) {
    int k = idx >> 6, n = idx & 63;
    sW[n * HID + k] = (bf16)W1[idx];
  }
  if (t < HID) { sMean[t] = mean[t]; sRstd[t] = rstd[t]; }
  __syncthreads();

  int wave = t >> 5, lane = t & 31;
  int tile_m = blockIdx.x * 32 + (wave >> 2) * 16;
  int tile_n = (wave & 3) * 16;
  int hi = lane >> 4;
  int lr = lane & 15;
  int row = tile_m + lr;
  int col = tile_n + lr;
  int kb  = hi * 8;

  v8f c = {0.f, 0.f, 0.f, 0.f, 0.f, 0.f, 0.f, 0.f};
  #pragma unroll
  for (int kk = 0; kk < 2; ++kk) {           // K = 64 in 2 steps of 32
    int kbase = kk * 32 + kb;
    const float* hr = h1 + (size_t)row * HID + kbase;
    float4 f0 = *(const float4*)(hr);
    float4 f1 = *(const float4*)(hr + 4);
    float4 f2 = *(const float4*)(hr + 16);
    float4 f3 = *(const float4*)(hr + 20);
    float fv[16];
    fv[0]=f0.x;  fv[1]=f0.y;  fv[2]=f0.z;  fv[3]=f0.w;
    fv[4]=f1.x;  fv[5]=f1.y;  fv[6]=f1.z;  fv[7]=f1.w;
    fv[8]=f2.x;  fv[9]=f2.y;  fv[10]=f2.z; fv[11]=f2.w;
    fv[12]=f3.x; fv[13]=f3.y; fv[14]=f3.z; fv[15]=f3.w;
    v16bf a;
    #pragma unroll
    for (int j = 0; j < 8; ++j) {
      int k0 = kbase + j, k1 = kbase + 16 + j;
      float v0 = (fv[j]     - sMean[k0]) * sRstd[k0];
      float v1 = (fv[8 + j] - sMean[k1]) * sRstd[k1];
      a[j]     = (bf16)fmaxf(v0, 0.0f);
      a[8 + j] = (bf16)fmaxf(v1, 0.0f);
    }
    const bf16* wc = &sW[col * HID + kbase];
    v8bf blo = *(const v8bf*)(wc);
    v8bf bhi = *(const v8bf*)(wc + 16);
    v16bf b = __builtin_shufflevector(blo, bhi, 0,1,2,3,4,5,6,7,8,9,10,11,12,13,14,15);
    c = __builtin_amdgcn_wmma_f32_16x16x32_bf16(false, a, false, b, (short)0, c,
                                                false, false);
  }
  float bias = b1[col];
  #pragma unroll
  for (int v = 0; v < 8; ++v)
    hout[(size_t)(tile_m + hi * 8 + v) * OUT_C + col] = c[v] + bias;
}

// ------------------------------ launcher -----------------------------------

extern "C" void kernel_launch(void* const* d_in, const int* in_sizes, int n_in,
                              void* d_out, int out_size, void* d_ws, size_t ws_size,
                              hipStream_t stream) {
  const float* x  = (const float*)d_in[0];
  const float* W0 = (const float*)d_in[1];
  const float* b0 = (const float*)d_in[2];
  const float* W1 = (const float*)d_in[3];
  const float* b1 = (const float*)d_in[4];
  const int*   ei = (const int*)d_in[5];

  const int N = in_sizes[0] / IN_C;   // 100000
  const int E = in_sizes[5] / 2;      // 1600000
  const int* src = ei;
  const int* dst = ei + E;
  float* out = (float*)d_out;

  // workspace layout (floats):
  //   [h1 / bufB : N*64][bufA : N*64][dinv : N][sum:64][sq:64][mean:64][rstd:64][w : E]
  float* ws   = (float*)d_ws;
  float* bufB = ws;                       // GEMM1 output, then ping-pong buffer
  float* bufA = bufB + (size_t)N * HID;   // GEMM2 output, then ping-pong buffer
  float* dinv = bufA + (size_t)N * OUT_C;
  float* gsum = dinv + N;
  float* gsq  = gsum + HID;
  float* mean = gsq + HID;
  float* rstd = mean + HID;
  float* wbuf = rstd + HID;

  const int eb  = (E + 255) / 256;
  const int nb  = (N + 255) / 256;
  const int gmb = N / 32;                 // 100000 / 32 = 3125 GEMM blocks

  // 1) zero dinv(N) + sum(64) + sumsq(64) (contiguous region)
  zero_f32<<<(N + 2 * HID + 255) / 256, 256, 0, stream>>>(dinv, N + 2 * HID);

  // 2) h1 = x@W0 + b0 (WMMA bf16) with fused BN stat accumulation
  gemm1_wmma<<<gmb, 256, 0, stream>>>(x, W0, b0, bufB, gsum, gsq, N);

  // 3) mean / rstd
  bn_finalize<<<1, 64, 0, stream>>>(gsum, gsq, mean, rstd, N);

  // 4-6) GCN normalization: degree -> dinv -> per-edge weights
  deg_kernel<<<eb, 256, 0, stream>>>(src, dst, dinv, E);
  dinv_kernel<<<nb, 256, 0, stream>>>(dinv, N);
  edgew_kernel<<<eb, 256, 0, stream>>>(src, dst, dinv, wbuf, E);

  // 7) bufA = relu(BN(h1)) @ W1 + b1 (WMMA bf16, BN+ReLU fused)
  gemm2_wmma<<<gmb, 256, 0, stream>>>(bufB, W1, b1, mean, rstd, bufA, N);

  // 8) 5 APPNP propagation steps, ping-pong A <-> B, last step into d_out
  const int ib = (N * 16 + 255) / 256;
  const int sb = (E * 16 + 255) / 256;
  float* pin[5]  = {bufA, bufB, bufA, bufB, bufA};
  float* pout[5] = {bufB, bufA, bufB, bufA, out};
  for (int s = 0; s < 5; ++s) {
    prop_init<<<ib, 256, 0, stream>>>(pin[s], dinv, pout[s], N);
    prop_scatter<<<sb, 256, 0, stream>>>(pin[s], pout[s], src, dst, wbuf, E);
  }
}
